// StaticInt8Linear_55164559949878
// MI455X (gfx1250) — 8-bit matrix operand order repaired and checked by reading, compiled, not run on hardware
//
#include <hip/hip_runtime.h>


typedef __attribute__((ext_vector_type(8))) int          v8i;
typedef __attribute__((ext_vector_type(4))) int          v4i;
typedef __attribute__((ext_vector_type(2))) int          v2i;
typedef __attribute__((ext_vector_type(8))) _Float16     v8h;
typedef __attribute__((ext_vector_type(4))) unsigned int u32x4;
typedef __attribute__((ext_vector_type(8))) int          i32x8;

constexpr int CI   = 1024;
constexpr int CO   = 1024;
constexpr int MTOT = 16 * 2048;

constexpr int BM = 128;
constexpr int BN = 64;
constexpr int KC = 64;
constexpr int NKI = CI / KC;

constexpr int ROWB = 80;
constexpr int ABUF = BM * ROWB;
constexpr int BBUF = BN * ROWB;
constexpr int BUFB = ABUF + BBUF;

__global__ __launch_bounds__(256) void quant_kernel(
    const _Float16* __restrict__ x,
    const float*    __restrict__ act_scale,
    signed char*    __restrict__ xq,
    int n8)
{
    int i = blockIdx.x * blockDim.x + threadIdx.x;
    if (i >= n8) return;
    const float inv = 1.0f / act_scale[0];

    v8h v = *(const v8h*)(x + (size_t)i * 8);

    unsigned long long packed = 0;
#pragma unroll
    for (int j = 0; j < 8; ++j) {
        float f = (float)v[j] * inv;
        f = __builtin_rintf(f);
        f = fminf(fmaxf(f, -128.0f), 127.0f);
        int q = (int)f;
        packed |= ((unsigned long long)(unsigned char)(signed char)q) << (8 * j);
    }
    *(unsigned long long*)(xq + (size_t)i * 8) = packed;
}

__device__ __forceinline__ void tdm_load_2d(unsigned lds_off,
                                            const void* gptr,
                                            unsigned tile_w_bytes,
                                            unsigned tile_h_rows,
                                            unsigned row_stride_bytes)
{
    unsigned long long ga = (unsigned long long)gptr;

    u32x4 g0;
    g0[0] = 1u;
    g0[1] = lds_off;
    g0[2] = (unsigned)(ga & 0xffffffffu);
    g0[3] = (unsigned)((ga >> 32) & 0x01ffffffu)
          | (2u << 30);

    const unsigned td0 = 1u << 30;
    const unsigned td1 = 1u << 30;
    i32x8 g1;
    g1[0] = (1 << 20)
          | (3 << 22)
          | (3 << 25);
    g1[1] = (int)((td0 & 0xffffu) << 16);
    g1[2] = (int)(((td0 >> 16) & 0xffffu) | ((td1 & 0xffffu) << 16));
    g1[3] = (int)(((td1 >> 16) & 0xffffu) | (tile_w_bytes << 16));
    g1[4] = (int)(tile_h_rows & 0xffffu);
    g1[5] = (int)row_stride_bytes;
    g1[6] = 0;
    g1[7] = 0;

    v4i  g2 = {0, 0, 0, 0};
    v4i  g3 = {0, 0, 0, 0};
    i32x8 g4 = {0, 0, 0, 0, 0, 0, 0, 0};

    __builtin_amdgcn_tensor_load_to_lds(g0, g1, g2, g3, g4,  0);
}

__global__ __launch_bounds__(256) void gemm_i8_kernel(
    const signed char* __restrict__ Aq,
    const signed char* __restrict__ W,
    const float*       __restrict__ wscale,
    const float*       __restrict__ act_scale,
    const float*       __restrict__ bias,
    _Float16*          __restrict__ out)
{
    const int lane = threadIdx.x & 31;
    const int wave = threadIdx.x >> 5;
    const int wm   = wave >> 1;
    const int wn   = wave & 1;
    const int lr   = lane & 15;
    const int lh   = lane >> 4;
    const int mBlk = blockIdx.x * BM;
    const int nBlk = blockIdx.y * BN;
    const int m0   = mBlk + wm * 32;
    const int n0   = nBlk + wn * 32;

    v8i acc[2][2] = {};

    __shared__ __align__(128) signed char smem[2 * BUFB];
    const unsigned ldsBase = (unsigned)(unsigned long long)(void*)&smem[0];

    if (wave == 0) {
        tdm_load_2d(ldsBase,        Aq + (size_t)mBlk * CI, KC, BM, CI);
        tdm_load_2d(ldsBase + ABUF, W  + (size_t)nBlk * CI, KC, BN, CI);
    }

    const int aRow = wm * 32 + lr;
    const int bRow = wn * 32 + lr;

    for (int it = 0; it < NKI; ++it) {
        const int buf = it & 1;

        if (wave == 0) {
            if (it + 1 < NKI) {
                const int k0 = (it + 1) * KC;
                const unsigned dst = ldsBase + (buf ^ 1) * BUFB;
                tdm_load_2d(dst,        Aq + (size_t)mBlk * CI + k0, KC, BM, CI);
                tdm_load_2d(dst + ABUF, W  + (size_t)nBlk * CI + k0, KC, BN, CI);
                __builtin_amdgcn_s_wait_tensorcnt(2);
            } else {
                __builtin_amdgcn_s_wait_tensorcnt(0);
            }
        }
        __syncthreads();

        v8i a[2], b[2];
#pragma unroll
        for (int mi = 0; mi < 2; ++mi) {
            const int base = buf * BUFB + (aRow + mi * 16) * ROWB + lh * 16;
            v2i t0 = *(const v2i*)&smem[base];
            v2i t1 = *(const v2i*)&smem[base + 8];
            v2i t2 = *(const v2i*)&smem[base + 32];
            v2i t3 = *(const v2i*)&smem[base + 40];
            a[mi][0] = t0[0]; a[mi][1] = t0[1];
            a[mi][2] = t1[0]; a[mi][3] = t1[1];
            a[mi][4] = t2[0]; a[mi][5] = t2[1];
            a[mi][6] = t3[0]; a[mi][7] = t3[1];
        }
#pragma unroll
        for (int ni = 0; ni < 2; ++ni) {
            const int base = buf * BUFB + ABUF + (bRow + ni * 16) * ROWB + lh * 16;
            v4i t0 = *(const v4i*)&smem[base];
            v4i t1 = *(const v4i*)&smem[base + 32];
            b[ni][0] = t0[0]; b[ni][1] = t0[1];
            b[ni][2] = t0[2]; b[ni][3] = t0[3];
            b[ni][4] = t1[0]; b[ni][5] = t1[1];
            b[ni][6] = t1[2]; b[ni][7] = t1[3];
        }

#pragma unroll
        for (int mi = 0; mi < 2; ++mi)
#pragma unroll
            for (int ni = 0; ni < 2; ++ni)
                acc[mi][ni] = __builtin_amdgcn_wmma_i32_16x16x64_iu8(
                    true, a[mi], true, b[ni], acc[mi][ni], false, false);

        __syncthreads();
    }

    const float as = act_scale[0];
#pragma unroll
    for (int ni = 0; ni < 2; ++ni) {
        const int   col = n0 + ni * 16 + lr;
        const float os  = wscale[col] * as;
        const float bs  = bias[col];
#pragma unroll
        for (int mi = 0; mi < 2; ++mi) {
#pragma unroll
            for (int r = 0; r < 8; ++r) {
                const int row = m0 + mi * 16 + lh * 8 + r;
                const float v = (float)acc[mi][ni][r] * os + bs;
                out[(size_t)row * CO + col] = (_Float16)v;
            }
        }
    }
}

extern "C" void kernel_launch(void* const* d_in, const int* in_sizes, int n_in,
                              void* d_out, int out_size, void* d_ws, size_t ws_size,
                              hipStream_t stream)
{
    const _Float16*    x      = (const _Float16*)d_in[0];
    const signed char* w      = (const signed char*)d_in[1];
    const float*       wscale = (const float*)d_in[2];
    const float*       ascale = (const float*)d_in[3];
    const float*       bias   = (const float*)d_in[4];
    _Float16*          out    = (_Float16*)d_out;
    signed char*       xq     = (signed char*)d_ws;

    const int n8 = MTOT * CI / 8;
    quant_kernel<<<n8 / 256, 256, 0, stream>>>(x, ascale, xq, n8);

    dim3 grid(MTOT / BM, CO / BN);
    gemm_i8_kernel<<<grid, 256, 0, stream>>>(xq, w, wscale, ascale, bias, out);
}
